// CFMM_52810917872265
// MI455X (gfx1250) — matrix operand order repaired and checked by reading, not compiled, not run on hardware
//
#include <hip/hip_runtime.h>
#include <math.h>

typedef __attribute__((ext_vector_type(16))) _Float16 v16h;
typedef __attribute__((ext_vector_type(8)))  float    v8f;

#define ATTN_SCALE 0.35355339059327373f  // 8^-0.5
#define EPSV 1e-5f

// ---------------------------------------------------------------------------
// Global min/max reduction (two stage)
// ---------------------------------------------------------------------------
__global__ void k_minmax_part(const float* __restrict__ in, int n,
                              float* __restrict__ part) {
    __shared__ float smn[256], smx[256];
    int tid = threadIdx.x;
    float mn = 3.4e38f, mx = -3.4e38f;
    for (int i = blockIdx.x * 256 + tid; i < n; i += 256 * gridDim.x) {
        float v = in[i];
        mn = fminf(mn, v); mx = fmaxf(mx, v);
    }
    smn[tid] = mn; smx[tid] = mx;
    __syncthreads();
    for (int s = 128; s > 0; s >>= 1) {
        if (tid < s) {
            smn[tid] = fminf(smn[tid], smn[tid + s]);
            smx[tid] = fmaxf(smx[tid], smx[tid + s]);
        }
        __syncthreads();
    }
    if (tid == 0) { part[2 * blockIdx.x] = smn[0]; part[2 * blockIdx.x + 1] = smx[0]; }
}

__global__ void k_minmax_final(const float* __restrict__ part, int nb,
                               float* __restrict__ stats, int slot) {
    __shared__ float smn[256], smx[256];
    int tid = threadIdx.x;
    float mn = 3.4e38f, mx = -3.4e38f;
    for (int i = tid; i < nb; i += 256) {
        mn = fminf(mn, part[2 * i]); mx = fmaxf(mx, part[2 * i + 1]);
    }
    smn[tid] = mn; smx[tid] = mx;
    __syncthreads();
    for (int s = 128; s > 0; s >>= 1) {
        if (tid < s) {
            smn[tid] = fminf(smn[tid], smn[tid + s]);
            smx[tid] = fmaxf(smx[tid], smx[tid + s]);
        }
        __syncthreads();
    }
    if (tid == 0) { stats[2 * slot] = smn[0]; stats[2 * slot + 1] = smx[0]; }
}

// ---------------------------------------------------------------------------
// FAM
// ---------------------------------------------------------------------------
__global__ void k_fam_mxav(const float* __restrict__ a, const float* __restrict__ b,
                           const float* __restrict__ stats, int sa, int sb,
                           int C, int HW, float* __restrict__ map2) {
    int idx = blockIdx.x * blockDim.x + threadIdx.x;
    if (idx >= 2 * HW) return;
    int bb = idx / HW, hw = idx % HW;
    float amin = stats[2 * sa], amax = stats[2 * sa + 1];
    float bmin = stats[2 * sb], bmax = stats[2 * sb + 1];
    float ari = 1.f / (amax - amin), bri = 1.f / (bmax - bmin);
    float fmx = -3.4e38f, fsum = 0.f;
    const float* ap = a + (size_t)bb * C * HW + hw;
    const float* bp = b + (size_t)bb * C * HW + hw;
    for (int c = 0; c < C; ++c) {
        float na = (ap[(size_t)c * HW] - amin) * ari;
        float nb = (bp[(size_t)c * HW] - bmin) * bri;
        float f = na * (1.f - nb);
        fmx = fmaxf(fmx, f); fsum += f;
    }
    map2[(size_t)(bb * 2 + 0) * HW + hw] = fmx;
    map2[(size_t)(bb * 2 + 1) * HW + hw] = fsum / (float)C;
}

__global__ void k_fam_apply(const float* __restrict__ a, const float* __restrict__ map2,
                            const float* __restrict__ w, const float* __restrict__ wb,
                            int C, int H, int W, float* __restrict__ out) {
    int idx = blockIdx.x * blockDim.x + threadIdx.x;
    int HW = H * W;
    if (idx >= 2 * HW) return;
    int bb = idx / HW, hw = idx % HW;
    int hh = hw / W, ww = hw % W;
    float acc = wb[0];
    for (int ch = 0; ch < 2; ++ch) {
        const float* mp = map2 + (size_t)(bb * 2 + ch) * HW;
        const float* wp = w + ch * 49;
        for (int ky = 0; ky < 7; ++ky) {
            int iy = hh + ky - 3; if (iy < 0 || iy >= H) continue;
            for (int kx = 0; kx < 7; ++kx) {
                int ix = ww + kx - 3; if (ix < 0 || ix >= W) continue;
                acc += mp[iy * W + ix] * wp[ky * 7 + kx];
            }
        }
    }
    float attn = 1.f / (1.f + __expf(-acc));
    const float* ap = a + (size_t)bb * C * HW + hw;
    float* op = out + (size_t)bb * C * HW + hw;
    for (int c = 0; c < C; ++c) op[(size_t)c * HW] = ap[(size_t)c * HW] * attn;
}

// ---------------------------------------------------------------------------
// Weight padding helpers (pad to K%32==0, N%32==0 once per launch)
// ---------------------------------------------------------------------------
__global__ void k_padw(const float* __restrict__ src, int N, int K,
                       float* __restrict__ dst, int Npad, int Kpad) {
    int idx = blockIdx.x * blockDim.x + threadIdx.x;
    if (idx >= Npad * Kpad) return;
    int n = idx / Kpad, k = idx % Kpad;
    dst[idx] = (n < N && k < K) ? src[n * K + k] : 0.f;
}

__global__ void k_padb(const float* __restrict__ src, int N,
                       float* __restrict__ dst, int Npad) {
    int idx = blockIdx.x * blockDim.x + threadIdx.x;
    if (idx >= Npad) return;
    dst[idx] = (idx < N) ? src[idx] : 0.f;
}

__global__ void k_fill0(float* __restrict__ p, int n) {
    int idx = blockIdx.x * blockDim.x + threadIdx.x;
    if (idx < n) p[idx] = 0.f;
}

// ---------------------------------------------------------------------------
// WMMA GEMM: Out[M,Npad] = A[M,Kpad] @ W[Npad,Kpad]^T (+bias).
// Requires Kpad%32==0, Npad%32==0, M%16==0. One wave = 16x32 tile (2 WMMAs/step).
// Branch-free: EXEC stays all-ones (WMMA requirement), loads are float4.
// ---------------------------------------------------------------------------
__device__ inline v16h pack16(float4 a, float4 b, float4 c, float4 d) {
    v16h r;
    r[0]  = (_Float16)a.x; r[1]  = (_Float16)a.y; r[2]  = (_Float16)a.z; r[3]  = (_Float16)a.w;
    r[4]  = (_Float16)b.x; r[5]  = (_Float16)b.y; r[6]  = (_Float16)b.z; r[7]  = (_Float16)b.w;
    r[8]  = (_Float16)c.x; r[9]  = (_Float16)c.y; r[10] = (_Float16)c.z; r[11] = (_Float16)c.w;
    r[12] = (_Float16)d.x; r[13] = (_Float16)d.y; r[14] = (_Float16)d.z; r[15] = (_Float16)d.w;
    return r;
}

__global__ __launch_bounds__(32)
void k_gemm(const float* __restrict__ A, const float* __restrict__ W,
            const float* __restrict__ bias, float* __restrict__ Out,
            int K, int N, int has_bias) {
    int lane = threadIdx.x;       // 0..31
    int half = lane >> 4;         // 0 or 1
    int l    = lane & 15;
    int tileM = blockIdx.x * 16;
    int tileN = blockIdx.y * 32;
    const float* arowp = A + (size_t)(tileM + l) * K;       // A row for this lane
    const float* b0p   = W + (size_t)(tileN + l) * K;       // weight row = out col
    const float* b1p   = W + (size_t)(tileN + 16 + l) * K;
    v8f acc0 = {}, acc1 = {};
    for (int k0 = 0; k0 < K; k0 += 32) {
        // A 16x32 f16 fragment: halves = 16 contiguous K at k0+16*half
        const float4* a0 = (const float4*)(arowp + k0 + 16 * half);
        v16h af = pack16(a0[0], a0[1], a0[2], a0[3]);
        // B 32x16 f16 fragment: lane = N, halves = 16 contiguous K at k0+16*half
        const float4* b0 = (const float4*)(b0p + k0 + 16 * half);
        const float4* b1 = (const float4*)(b1p + k0 + 16 * half);
        v16h bf0 = pack16(b0[0], b0[1], b0[2], b0[3]);
        v16h bf1 = pack16(b1[0], b1[1], b1[2], b1[3]);
        acc0 = __builtin_amdgcn_wmma_f32_16x16x32_f16(false, af, false, bf0,
                                                      (short)0, acc0, false, false);
        acc1 = __builtin_amdgcn_wmma_f32_16x16x32_f16(false, af, false, bf1,
                                                      (short)0, acc1, false, false);
    }
    int oc0 = tileN + l, oc1 = tileN + 16 + l;
    float bv0 = has_bias ? bias[oc0] : 0.f;
    float bv1 = has_bias ? bias[oc1] : 0.f;
#pragma unroll
    for (int r = 0; r < 8; ++r) {
        int orow = tileM + r + 8 * half;   // C/D layout: lanes 16-31 hold rows 8..15
        Out[(size_t)orow * N + oc0] = acc0[r] + bv0;
        Out[(size_t)orow * N + oc1] = acc1[r] + bv1;
    }
}

// ---------------------------------------------------------------------------
// HiLo pieces (padded strides: qkv 32, oh/hifi 32, q/olo/lofi 64, kv 128)
// ---------------------------------------------------------------------------
__global__ void k_layernorm(const float* __restrict__ in, const float* __restrict__ nw,
                            const float* __restrict__ nb, float* __restrict__ xn) {
    int idx = blockIdx.x * blockDim.x + threadIdx.x;
    if (idx >= 6272) return;
    int b = idx / 3136, hw = idx % 3136;
    const float* ip = in + (size_t)b * 64 * 3136 + hw;
    float mu = 0.f;
    for (int c = 0; c < 64; ++c) mu += ip[(size_t)c * 3136];
    mu *= (1.f / 64.f);
    float var = 0.f;
    for (int c = 0; c < 64; ++c) { float d = ip[(size_t)c * 3136] - mu; var += d * d; }
    var *= (1.f / 64.f);
    float rs = rsqrtf(var + EPSV);
    float* op = xn + (size_t)idx * 64;
    for (int c = 0; c < 64; ++c)
        op[c] = (ip[(size_t)c * 3136] - mu) * rs * nw[c] + nb[c];
}

__global__ void k_hifi(const float* __restrict__ qkv, float* __restrict__ oh) {
    // qkv [6272,32] (q 0..7, k 8..15, v 16..23); 2x2 window, 1 head, d=8
    int wid = blockIdx.x * blockDim.x + threadIdx.x;
    if (wid >= 1568) return;
    int b = wid / 784, g = wid % 784;
    int hg = g / 28, wg = g % 28;
    int pix[4];
#pragma unroll
    for (int t = 0; t < 4; ++t) {
        int ii = t >> 1, jj = t & 1;
        pix[t] = b * 3136 + (2 * hg + ii) * 56 + (2 * wg + jj);
    }
    float q[4][8], k[4][8], v[4][8];
#pragma unroll
    for (int t = 0; t < 4; ++t) {
        const float* p = qkv + (size_t)pix[t] * 32;
#pragma unroll
        for (int d = 0; d < 8; ++d) { q[t][d] = p[d]; k[t][d] = p[8 + d]; v[t][d] = p[16 + d]; }
    }
#pragma unroll
    for (int i = 0; i < 4; ++i) {
        float s[4];
        float mx = -3.4e38f;
#pragma unroll
        for (int j = 0; j < 4; ++j) {
            float acc = 0.f;
#pragma unroll
            for (int d = 0; d < 8; ++d) acc += q[i][d] * k[j][d];
            s[j] = acc * ATTN_SCALE;
            mx = fmaxf(mx, s[j]);
        }
        float sum = 0.f;
#pragma unroll
        for (int j = 0; j < 4; ++j) { s[j] = __expf(s[j] - mx); sum += s[j]; }
        float rinv = 1.f / sum;
        float* op = oh + (size_t)pix[i] * 32;
#pragma unroll
        for (int d = 0; d < 8; ++d) {
            float acc = 0.f;
#pragma unroll
            for (int j = 0; j < 4; ++j) acc += s[j] * v[j][d];
            op[d] = acc * rinv;
        }
    }
}

__global__ void k_pool(const float* __restrict__ xn, float* __restrict__ xp) {
    int idx = blockIdx.x * blockDim.x + threadIdx.x;
    if (idx >= 1568 * 64) return;
    int c = idx % 64, bg = idx / 64;
    int b = bg / 784, g = bg % 784;
    int hg = g / 28, wg = g % 28;
    float s = 0.f;
#pragma unroll
    for (int ii = 0; ii < 2; ++ii)
#pragma unroll
        for (int jj = 0; jj < 2; ++jj) {
            int pix = b * 3136 + (2 * hg + ii) * 56 + (2 * wg + jj);
            s += xn[(size_t)pix * 64 + c];
        }
    xp[(size_t)bg * 64 + c] = s * 0.25f;
}

__global__ void k_lofi(const float* __restrict__ q, const float* __restrict__ kv,
                       float* __restrict__ olo) {
    // q [6272,64] ([head,dim] in cols 0..55), kv [1568,128] (k 0..55, v 56..111)
    int tid = blockIdx.x * blockDim.x + threadIdx.x;
    if (tid >= 2 * 7 * 3136) return;
    int b = tid / (7 * 3136);
    int r = tid % (7 * 3136);
    int h = r / 3136, i = r % 3136;
    const float* qp = q + (size_t)(b * 3136 + i) * 64 + h * 8;
    float qr[8];
#pragma unroll
    for (int d = 0; d < 8; ++d) qr[d] = qp[d];
    const float* kvb = kv + (size_t)b * 784 * 128;
    float m = -3.4e38f, l = 0.f, acc[8];
#pragma unroll
    for (int d = 0; d < 8; ++d) acc[d] = 0.f;
    for (int j = 0; j < 784; ++j) {
        const float* kp = kvb + (size_t)j * 128 + h * 8;
        float s = 0.f;
#pragma unroll
        for (int d = 0; d < 8; ++d) s += qr[d] * kp[d];
        s *= ATTN_SCALE;
        float mn = fmaxf(m, s);
        float eo = __expf(m - mn);
        float en = __expf(s - mn);
        l = l * eo + en;
        const float* vp = kp + 56;
#pragma unroll
        for (int d = 0; d < 8; ++d) acc[d] = acc[d] * eo + en * vp[d];
        m = mn;
    }
    float rinv = 1.f / l;
    float* op = olo + (size_t)(b * 3136 + i) * 64 + h * 8;
#pragma unroll
    for (int d = 0; d < 8; ++d) op[d] = acc[d] * rinv;
}

__global__ void k_combine(const float* __restrict__ inp, const float* __restrict__ hifi,
                          const float* __restrict__ lofi, float* __restrict__ out) {
    int idx = blockIdx.x * blockDim.x + threadIdx.x;
    if (idx >= 2 * 64 * 3136) return;
    int hw = idx % 3136;
    int c = (idx / 3136) % 64;
    int b = idx / (64 * 3136);
    int pix = b * 3136 + hw;
    float v = (c < 8) ? hifi[(size_t)pix * 32 + c] : lofi[(size_t)pix * 64 + (c - 8)];
    out[idx] = inp[idx] + v;
}

// ---------------------------------------------------------------------------
// Tail
// ---------------------------------------------------------------------------
__global__ void k_up2(const float* __restrict__ in, int C, float* __restrict__ out) {
    int idx = blockIdx.x * blockDim.x + threadIdx.x;
    if (idx >= 2 * C * 3136) return;
    int hw = idx % 3136;
    int c = (idx / 3136) % C;
    int b = idx / (3136 * C);
    int ho = hw / 56, wo = hw % 56;
    float sy = ho * 0.5f - 0.25f;
    float sx = wo * 0.5f - 0.25f;
    int y0 = (int)floorf(sy), x0 = (int)floorf(sx);
    float fy = sy - (float)y0, fx = sx - (float)x0;
    int y0c = min(max(y0, 0), 27), y1c = min(max(y0 + 1, 0), 27);
    int x0c = min(max(x0, 0), 27), x1c = min(max(x0 + 1, 0), 27);
    const float* p = in + (size_t)(b * C + c) * 784;
    float v = (1.f - fy) * ((1.f - fx) * p[y0c * 28 + x0c] + fx * p[y0c * 28 + x1c]) +
              fy * ((1.f - fx) * p[y1c * 28 + x0c] + fx * p[y1c * 28 + x1c]);
    out[idx] = v;
}

__global__ void k_conv1x1_mul(const float* __restrict__ xin, const float* __restrict__ w,
                              const float* __restrict__ bias, const float* __restrict__ mul,
                              float* __restrict__ out) {
    int idx = blockIdx.x * blockDim.x + threadIdx.x;
    if (idx >= 2 * 64 * 3136) return;
    int hw = idx % 3136;
    int co = (idx / 3136) % 64;
    int b = idx / (3136 * 64);
    const float* ip = xin + (size_t)b * 128 * 3136 + hw;
    const float* wp = w + co * 128;
    float s = bias[co];
    for (int ci = 0; ci < 128; ++ci) s += wp[ci] * ip[(size_t)ci * 3136];
    out[idx] = mul[idx] * s;
}

__global__ void k_concat(const float* __restrict__ a, const float* __restrict__ b2,
                         float* __restrict__ out) {
    int idx = blockIdx.x * blockDim.x + threadIdx.x;
    if (idx >= 2 * 128 * 3136) return;
    int hw = idx % 3136;
    int c = (idx / 3136) % 128;
    int b = idx / (3136 * 128);
    out[idx] = (c < 64) ? a[(size_t)(b * 64 + c) * 3136 + hw]
                        : b2[(size_t)(b * 64 + (c - 64)) * 3136 + hw];
}

__global__ void k_conv3_bn_relu(const float* __restrict__ in, int Cin,
                                const float* __restrict__ w, const float* __restrict__ cb,
                                const float* __restrict__ g, const float* __restrict__ bb,
                                const float* __restrict__ m, const float* __restrict__ v,
                                int Cout, float* __restrict__ out) {
    int idx = blockIdx.x * blockDim.x + threadIdx.x;
    if (idx >= 2 * Cout * 3136) return;
    int hw = idx % 3136;
    int co = (idx / 3136) % Cout;
    int b = idx / (3136 * Cout);
    int hh = hw / 56, ww = hw % 56;
    float s = cb[co];
    for (int ci = 0; ci < Cin; ++ci) {
        const float* ip = in + (size_t)(b * Cin + ci) * 3136;
        const float* wp = w + (size_t)(co * Cin + ci) * 9;
#pragma unroll
        for (int ky = 0; ky < 3; ++ky) {
            int iy = hh + ky - 1; if (iy < 0 || iy >= 56) continue;
#pragma unroll
            for (int kx = 0; kx < 3; ++kx) {
                int ix = ww + kx - 1; if (ix < 0 || ix >= 56) continue;
                s += ip[iy * 56 + ix] * wp[ky * 3 + kx];
            }
        }
    }
    float y = (s - m[co]) * rsqrtf(v[co] + EPSV) * g[co] + bb[co];
    out[idx] = fmaxf(y, 0.f);
}

__global__ void k_proj2(const float* __restrict__ in, const float* __restrict__ w,
                        const float* __restrict__ bias, float* __restrict__ out) {
    int idx = blockIdx.x * blockDim.x + threadIdx.x;
    if (idx >= 2 * 3136) return;
    int b = idx / 3136, hw = idx % 3136;
    float s = bias[0];
    const float* ip = in + (size_t)b * 32 * 3136 + hw;
    for (int c = 0; c < 32; ++c) s += w[c] * ip[(size_t)c * 3136];
    out[idx] = s;
}

// ---------------------------------------------------------------------------
// Driver
// ---------------------------------------------------------------------------
extern "C" void kernel_launch(void* const* d_in, const int* in_sizes, int n_in,
                              void* d_out, int out_size, void* d_ws, size_t ws_size,
                              hipStream_t stream) {
    auto P = [&](int i) -> const float* { return (const float*)d_in[i]; };
    const float* x1 = P(0);  // [2,64,56,56]
    const float* y1 = P(1);
    const float* x2 = P(2);  // [2,128,28,28]
    const float* y2 = P(3);
    // d_in[4] = size (unused); params 5..140 in _make_params() dict order

    float* WSF = (float*)d_ws;
    const size_t O_STATS = 0;
    const size_t O_PART  = 16;
    const size_t O_X2F   = 528;
    const size_t O_Y2F   = O_X2F + 200704;
    const size_t O_XA    = O_Y2F + 200704;
    const size_t O_XB    = O_XA + 401408;
    const size_t O_YA    = O_XB + 401408;
    const size_t O_YB    = O_YA + 401408;
    const size_t O_TMP   = O_YB + 401408;

    float* stats = WSF + O_STATS;
    float* part  = WSF + O_PART;
    float* X2F   = WSF + O_X2F;
    float* Y2F   = WSF + O_Y2F;
    float* XA    = WSF + O_XA;
    float* XB    = WSF + O_XB;
    float* YA    = WSF + O_YA;
    float* YB    = WSF + O_YB;
    // HiLo temps (alive only during HiLo phase) — padded strides
    float* xn   = WSF + O_TMP;            // [6272,64]   401408
    float* qkvh = xn + 401408;            // [6272,32]   200704
    float* oh   = qkvh + 200704;          // [6272,32]   200704
    float* hifi = oh + 200704;            // [6272,32]   200704
    float* xp   = hifi + 200704;          // [1568,64]   100352
    float* kvb  = xp + 100352;            // [1568,128]  200704
    float* qb   = kvb + 200704;           // [6272,64]   401408
    float* olo  = qb + 401408;            // [6272,64]   401408
    float* lofi = olo + 401408;           // [6272,64]   401408
    float* wpad = lofi + 401408;          // 24 * 19552 = 469248
    const size_t WPB = 19552;             // per-block padded weights:
    // +0 hqkv[32,64], +2048 hproj[32,32], +3072 hprojb[32],
    // +3104 lq[64,64], +7200 lkv[128,64], +15392 lproj[64,64], +19488 lprojb[64]
    // Post-stage buffers (alias HiLo temp region — HiLo done by then)
    float* x2u = WSF + O_TMP;             // 802816
    float* y2u = x2u + 802816;            // 802816
    float* fx  = y2u + 802816;            // 401408
    float* fy  = fx + 401408;             // 401408
    float* cat = fy + 401408;             // 802816
    float* h1  = cat + 802816;            // 401408
    float* h3  = h1 + 401408;             // 200704
    float* map2 = WSF + O_TMP;            // FAM-phase scratch (<=12544)

    float* feats   = (float*)d_out;            // [2,32,56,56]
    float* predict = (float*)d_out + 200704;   // [2,1,56,56]

    dim3 B256(256);
    auto GB = [](int n) { return dim3((n + 255) / 256); };

    // ---- global min/max of the four inputs ----
    k_minmax_part<<<256, B256, 0, stream>>>(x1, 401408, part);
    k_minmax_final<<<1, B256, 0, stream>>>(part, 256, stats, 0);
    k_minmax_part<<<256, B256, 0, stream>>>(y1, 401408, part);
    k_minmax_final<<<1, B256, 0, stream>>>(part, 256, stats, 1);
    k_minmax_part<<<256, B256, 0, stream>>>(x2, 200704, part);
    k_minmax_final<<<1, B256, 0, stream>>>(part, 256, stats, 2);
    k_minmax_part<<<256, B256, 0, stream>>>(y2, 200704, part);
    k_minmax_final<<<1, B256, 0, stream>>>(part, 256, stats, 3);

    // ---- FAM gates (fam2 at 5/6, fam1 at 7/8) ----
    k_fam_mxav<<<GB(2 * 784), B256, 0, stream>>>(x2, y2, stats, 2, 3, 128, 784, map2);
    k_fam_apply<<<GB(2 * 784), B256, 0, stream>>>(x2, map2, P(5), P(6), 128, 28, 28, X2F);
    k_fam_mxav<<<GB(2 * 784), B256, 0, stream>>>(y2, x2, stats, 3, 2, 128, 784, map2);
    k_fam_apply<<<GB(2 * 784), B256, 0, stream>>>(y2, map2, P(5), P(6), 128, 28, 28, Y2F);
    k_fam_mxav<<<GB(2 * 3136), B256, 0, stream>>>(x1, y1, stats, 0, 1, 64, 3136, map2);
    k_fam_apply<<<GB(2 * 3136), B256, 0, stream>>>(x1, map2, P(7), P(8), 64, 56, 56, XA);
    k_fam_mxav<<<GB(2 * 3136), B256, 0, stream>>>(y1, x1, stats, 1, 0, 64, 3136, map2);
    k_fam_apply<<<GB(2 * 3136), B256, 0, stream>>>(y1, map2, P(7), P(8), 64, 56, 56, YA);

    // ---- pad all HiLo weights once (N,K -> multiples of 32) ----
    for (int blk = 0; blk < 24; ++blk) {
        int base = (blk < 12) ? (25 + 9 * blk) : (79 + 9 * (blk - 12));
        float* wb = wpad + (size_t)blk * WPB;
        k_padw<<<GB(2048), B256, 0, stream>>>(P(base + 6), 24, 64, wb + 0,     32, 64);
        k_padw<<<GB(1024), B256, 0, stream>>>(P(base + 7),  8,  8, wb + 2048,  32, 32);
        k_padb<<<GB(32),   B256, 0, stream>>>(P(base + 8),  8,    wb + 3072,  32);
        k_padw<<<GB(4096), B256, 0, stream>>>(P(base + 2), 56, 64, wb + 3104,  64, 64);
        k_padw<<<GB(8192), B256, 0, stream>>>(P(base + 3), 112, 64, wb + 7200, 128, 64);
        k_padw<<<GB(4096), B256, 0, stream>>>(P(base + 4), 56, 56, wb + 15392, 64, 64);
        k_padb<<<GB(64),   B256, 0, stream>>>(P(base + 5), 56,    wb + 19488, 64);
    }
    // zero the padded activation columns once (stay zero across all blocks)
    k_fill0<<<GB(200704), B256, 0, stream>>>(oh, 200704);
    k_fill0<<<GB(401408), B256, 0, stream>>>(olo, 401408);

    // ---- HiLo block ----
    auto run_hilo = [&](const float* inbuf, float* outbuf, int blk) {
        int base = (blk < 12) ? (25 + 9 * blk) : (79 + 9 * (blk - 12));
        const float* wb = wpad + (size_t)blk * WPB;
        k_layernorm<<<GB(6272), B256, 0, stream>>>(inbuf, P(base + 0), P(base + 1), xn);
        k_gemm<<<dim3(392, 1), 32, 0, stream>>>(xn, wb + 0, nullptr, qkvh, 64, 32, 0);
        k_hifi<<<GB(1568), B256, 0, stream>>>(qkvh, oh);
        k_gemm<<<dim3(392, 1), 32, 0, stream>>>(oh, wb + 2048, wb + 3072, hifi, 32, 32, 1);
        k_pool<<<GB(1568 * 64), B256, 0, stream>>>(xn, xp);
        k_gemm<<<dim3(98, 4), 32, 0, stream>>>(xp, wb + 7200, nullptr, kvb, 64, 128, 0);
        k_gemm<<<dim3(392, 2), 32, 0, stream>>>(xn, wb + 3104, nullptr, qb, 64, 64, 0);
        k_lofi<<<GB(2 * 7 * 3136), B256, 0, stream>>>(qb, kvb, olo);
        k_gemm<<<dim3(392, 2), 32, 0, stream>>>(olo, wb + 15392, wb + 19488, lofi, 64, 64, 1);
        k_combine<<<GB(401408), B256, 0, stream>>>(inbuf, hifi, lofi, outbuf);
    };

    const float* xc = XA; float* xnx = XB;
    for (int i = 0; i < 6; ++i) {
        run_hilo(xc, xnx, i);
        const float* t = xnx; xnx = (float*)xc; xc = t;
    }
    const float* yc = YA; float* ynx = YB;
    for (int i = 0; i < 6; ++i) {
        run_hilo(yc, ynx, 12 + i);
        const float* t = ynx; ynx = (float*)yc; yc = t;
    }

    // ---- tail ----
    k_up2<<<GB(2 * 128 * 3136), B256, 0, stream>>>(X2F, 128, x2u);
    k_up2<<<GB(2 * 128 * 3136), B256, 0, stream>>>(Y2F, 128, y2u);
    k_conv1x1_mul<<<GB(401408), B256, 0, stream>>>(x2u, P(9), P(10), xc, fx);
    k_conv1x1_mul<<<GB(401408), B256, 0, stream>>>(y2u, P(11), P(12), yc, fy);
    k_concat<<<GB(802816), B256, 0, stream>>>(fx, fy, cat);
    k_conv3_bn_relu<<<GB(401408), B256, 0, stream>>>(cat, 128, P(13), P(14),
        P(15), P(16), P(17), P(18), 64, h1);
    k_conv3_bn_relu<<<GB(200704), B256, 0, stream>>>(h1, 64, P(19), P(20),
        P(21), P(22), P(23), P(24), 32, feats);
    k_conv3_bn_relu<<<GB(200704), B256, 0, stream>>>(feats, 32, P(133), P(134),
        P(135), P(136), P(137), P(138), 32, h3);
    k_proj2<<<GB(6272), B256, 0, stream>>>(h3, P(139), P(140), predict);
}
